// RnnNet_64476049048269
// MI455X (gfx1250) — matrix operand order repaired and checked by reading, not compiled, not run on hardware
//
#include <hip/hip_runtime.h>
#include <hip/hip_bf16.h>
#include <cstdint>
#include <cstddef>

// Problem constants (match reference)
constexpr int B_BATCH = 100000;
constexpr int SEQ     = 28;
constexpr int K_IN    = 28;   // input features (padded to 32 for WMMA K)
constexpr int HID     = 64;
constexpr int G3      = 3 * HID;  // 192 gate rows (r,z,n)
constexpr int N_OUT   = 10;

typedef __attribute__((ext_vector_type(16))) __bf16 v16bf;
typedef __attribute__((ext_vector_type(8)))  __bf16 v8bf;
typedef __attribute__((ext_vector_type(8)))  float  v8f;
typedef int v4i_vs __attribute__((vector_size(4 * sizeof(int))));

static __device__ __forceinline__ __bf16 f2bf(float f) {
  // round-to-nearest-even f32 -> bf16
  uint32_t u = __builtin_bit_cast(uint32_t, f);
  uint32_t r = (u + 0x7FFFu + ((u >> 16) & 1u)) >> 16;
  return __builtin_bit_cast(__bf16, (uint16_t)r);
}

static __device__ __forceinline__ float bf2f(__bf16 b) {
  uint32_t u = ((uint32_t)__builtin_bit_cast(uint16_t, b)) << 16;
  return __builtin_bit_cast(float, u);
}

static __device__ __forceinline__ float sigmoid_fast(float x) {
  return __builtin_amdgcn_rcpf(1.0f + __expf(-x));
}

static __device__ __forceinline__ float tanh_fast(float x) {
  // tanh(x) = 1 - 2/(exp(2x)+1); saturates correctly at +-inf
  return 1.0f - 2.0f * __builtin_amdgcn_rcpf(1.0f + __expf(2.0f * x));
}

static __device__ __forceinline__ v8f wmma_bf16(v16bf a, v16bf b, v8f c) {
  // (neg_a, A, neg_b, B, c_mod, C, reuse_a, reuse_b)
  return __builtin_amdgcn_wmma_f32_16x16x32_bf16(false, a, false, b, (short)0, c,
                                                 false, false);
}

// ---- CDNA5 async global->LDS copy (ASYNCcnt-tracked DMA) ----
static __device__ __forceinline__ void async_load_b128(const void* g, void* l) {
#if __has_builtin(__builtin_amdgcn_global_load_async_to_lds_b128)
  __builtin_amdgcn_global_load_async_to_lds_b128(
      (__attribute__((address_space(1))) v4i_vs*)g,
      (__attribute__((address_space(3))) v4i_vs*)l, /*offset=*/0, /*cpol=*/0);
#else
  unsigned long long ga = (unsigned long long)g;
  unsigned int la =
      (unsigned int)(unsigned long long)(__attribute__((address_space(3))) char*)l;
  asm volatile("global_load_async_to_lds_b128 %0, %1, off"
               :: "v"(la), "v"(ga) : "memory");
#endif
}

static __device__ __forceinline__ void wait_async0() {
#if __has_builtin(__builtin_amdgcn_s_wait_asynccnt)
  __builtin_amdgcn_s_wait_asynccnt(0);
#else
  asm volatile("s_wait_asynccnt 0x0");
#endif
  asm volatile("" ::: "memory");  // keep LDS reads below the wait
}

__global__ __launch_bounds__(128) void gru_wmma_kernel(
    const float* __restrict__ x,    // [B, SEQ, K_IN]
    const float* __restrict__ Wih,  // [G3, K_IN]
    const float* __restrict__ Whh,  // [G3, HID]
    const float* __restrict__ bih,  // [G3]
    const float* __restrict__ bhh,  // [G3]
    const float* __restrict__ Wout, // [N_OUT, HID]
    const float* __restrict__ bout, // [N_OUT]
    float* __restrict__ out)        // [B, N_OUT]
{
  // Shared weights (bf16, K padded for W_ih) + per-wave staging buffers.
  __shared__ __align__(32) __bf16 sWih[G3 * 32];   // [g][k0..31], k>=28 zero
  __shared__ __align__(32) __bf16 sWhh[G3 * HID];  // [g][k0..63]
  __shared__ float sbih[G3];
  __shared__ float sbhh[G3];
  __shared__ float sWout[N_OUT * HID];
  __shared__ float sbout[N_OUT];
  // Double-buffered f32 x tiles, filled by async DMA (row stride 32 floats;
  // cols 28..31 are zero padding that the DMA never writes).
  __shared__ __align__(32) float  sXF[4][2][16 * 32];
  __shared__ __align__(32) __bf16 sH[4][16 * HID];  // per-wave hidden tile

  const int tid = threadIdx.x;

  // ---- cooperative weight staging ----
  for (int i = tid; i < G3 * 32; i += 128) {
    int g = i >> 5, k = i & 31;
    sWih[i] = f2bf(k < K_IN ? Wih[g * K_IN + k] : 0.0f);
  }
  for (int i = tid; i < G3 * HID; i += 128) {
    int g = i >> 6, k = i & 63;
    sWhh[i] = f2bf(Whh[g * HID + k]);
  }
  for (int i = tid; i < G3; i += 128) {
    sbih[i] = bih[i];
    sbhh[i] = bhh[i];
  }
  for (int i = tid; i < N_OUT * HID; i += 128) sWout[i] = Wout[i];
  if (tid < N_OUT) sbout[tid] = bout[tid];
  __syncthreads();

  const int wv   = tid >> 5;
  const int lane = tid & 31;
  const long tile = (long)blockIdx.x * 4 + wv;
  const int  base = (int)(tile * 16);  // first batch row of this wave's tile
  if (base >= B_BATCH) return;         // no further block-level syncs below

  float*  xb[2] = { sXF[wv][0], sXF[wv][1] };
  __bf16* ht    = sH[wv];

  const int col  = lane & 15;   // N (and A-row) index
  const int half = lane >> 4;   // lane-half selects K sub-range
  const int rb   = half * 8;    // C-layout row base
  const int a0   = half * 16;       // A chunk 0 K-base
  const int a1   = half * 16 + 8;   // A chunk 1 K-base

  // zero-init hidden state (h0 = 0)
  for (int i = lane; i < 16 * HID; i += 32)
    ht[i] = __builtin_bit_cast(__bf16, (uint16_t)0);

  // Zero only the regions the DMA never writes: pad cols 28..31 of every row,
  // and whole rows whose batch index is out of range (keeps NaN-free tiles).
  if (lane < 16) {
    const int r = lane;
    const bool oob = (base + r) >= B_BATCH;
#pragma unroll
    for (int bsel = 0; bsel < 2; bsel++) {
      if (oob) {
        for (int c = 0; c < 32; c++) xb[bsel][r * 32 + c] = 0.0f;
      } else {
#pragma unroll
        for (int c = 28; c < 32; c++) xb[bsel][r * 32 + c] = 0.0f;
      }
    }
  }

  // ---- per-lane async-DMA chunk plan: 16 rows x 7 x (16B) chunks = 112 ----
  const float* gsrc[4];
  float*       ldst[2][4];
  bool         act[4];
#pragma unroll
  for (int i = 0; i < 4; i++) {
    int c = lane + 32 * i;
    int r = c / 7;          // tile row
    int k = c % 7;          // 16B chunk within row (cols k*4 .. k*4+3)
    act[i]  = (c < 112) && ((base + r) < B_BATCH);
    gsrc[i] = x + (size_t)(base + r) * (SEQ * K_IN) + (size_t)k * 4;
    ldst[0][i] = xb[0] + r * 32 + k * 4;
    ldst[1][i] = xb[1] + r * 32 + k * 4;
  }

  // kick off t=0 prefetch
#pragma unroll
  for (int i = 0; i < 4; i++)
    if (act[i]) async_load_b128(gsrc[i], ldst[0][i]);

  // per-lane gate biases (depend only on column)
  float brz[4], bzz[4], binn[4], bhnn[4];
#pragma unroll
  for (int j = 0; j < 4; j++) {
    int c = j * 16 + col;
    brz[j]  = sbih[c] + sbhh[c];                    // r gate: i_r + h_r bias
    bzz[j]  = sbih[HID + c] + sbhh[HID + c];        // z gate
    binn[j] = sbih[2 * HID + c];                    // i_n bias
    bhnn[j] = sbhh[2 * HID + c];                    // h_n bias
  }

  // hidden state in f32 C-layout registers
  v8f hC[4];
#pragma unroll
  for (int j = 0; j < 4; j++)
#pragma unroll
    for (int v = 0; v < 8; v++) hC[j][v] = 0.0f;

  for (int t = 0; t < SEQ; t++) {
    // wait for this step's x tile (only this step's loads are outstanding)
    wait_async0();

    // build A fragment for x from the f32 LDS tile (convert to bf16)
    const float* xrow = xb[t & 1] + (size_t)col * 32;
    v8f xlo = *(const v8f*)(xrow + a0);
    v8f xhi = *(const v8f*)(xrow + a1);
    v16bf ax;
#pragma unroll
    for (int i = 0; i < 8; i++) {
      ax[i]     = f2bf(xlo[i]);
      ax[8 + i] = f2bf(xhi[i]);
    }

    // prefetch next step's x tile into the other buffer (overlaps the WMMAs)
    if (t + 1 < SEQ) {
      const int nb = (t + 1) & 1;
#pragma unroll
      for (int i = 0; i < 4; i++)
        if (act[i]) async_load_b128(gsrc[i] + (size_t)(t + 1) * K_IN, ldst[nb][i]);
    }

    // accumulators initialized with biases
    v8f ar[4], az[4], ain[4], ahn[4];
#pragma unroll
    for (int j = 0; j < 4; j++)
#pragma unroll
      for (int v = 0; v < 8; v++) {
        ar[j][v]  = brz[j];
        az[j][v]  = bzz[j];
        ain[j][v] = binn[j];
        ahn[j][v] = bhnn[j];
      }

    // ---- gi = x_t @ W_ih^T  (12 WMMAs, K=32 padded) ----
#pragma unroll
    for (int j = 0; j < 4; j++) {
      const __bf16* wp;
      wp = sWih + (size_t)(j * 16 + col) * 32 + half * 16;
      ar[j] = wmma_bf16(ax, *(const v16bf*)wp, ar[j]);
      wp = sWih + (size_t)(HID + j * 16 + col) * 32 + half * 16;
      az[j] = wmma_bf16(ax, *(const v16bf*)wp, az[j]);
      wp = sWih + (size_t)(2 * HID + j * 16 + col) * 32 + half * 16;
      ain[j] = wmma_bf16(ax, *(const v16bf*)wp, ain[j]);
    }

    // ---- gh = h @ W_hh^T  (24 WMMAs, K=64 in two chunks) ----
#pragma unroll
    for (int kb = 0; kb < HID; kb += 32) {
      v8bf hlo = *(const v8bf*)(ht + col * HID + kb + a0);
      v8bf hhi = *(const v8bf*)(ht + col * HID + kb + a1);
      v16bf ah = __builtin_shufflevector(hlo, hhi, 0, 1, 2, 3, 4, 5, 6, 7, 8, 9,
                                         10, 11, 12, 13, 14, 15);
#pragma unroll
      for (int j = 0; j < 4; j++) {
        const __bf16* wp;
        wp = sWhh + (size_t)(j * 16 + col) * HID + kb + half * 16;
        ar[j] = wmma_bf16(ah, *(const v16bf*)wp, ar[j]);
        wp = sWhh + (size_t)(HID + j * 16 + col) * HID + kb + half * 16;
        az[j] = wmma_bf16(ah, *(const v16bf*)wp, az[j]);
        wp = sWhh + (size_t)(2 * HID + j * 16 + col) * HID + kb + half * 16;
        ahn[j] = wmma_bf16(ah, *(const v16bf*)wp, ahn[j]);
      }
    }

    // ---- gates + state update (f32), mirror h to LDS as bf16 ----
#pragma unroll
    for (int j = 0; j < 4; j++) {
#pragma unroll
      for (int v = 0; v < 8; v++) {
        float r  = sigmoid_fast(ar[j][v]);
        float z  = sigmoid_fast(az[j][v]);
        float n  = tanh_fast(ain[j][v] + r * ahn[j][v]);
        float hn = (1.0f - z) * n + z * hC[j][v];
        hC[j][v] = hn;
        ht[(size_t)(rb + v) * HID + j * 16 + col] = f2bf(hn);
      }
    }
  }

  // ---- output projection: out = h @ W_out^T + b_out  (tiny, VALU) ----
#pragma unroll
  for (int q = 0; q < 5; q++) {
    int p = lane + 32 * q;            // 160 = 16 rows * 10 outs
    int r = p / N_OUT, o = p % N_OUT;
    int b = base + r;
    if (b < B_BATCH) {
      float acc = sbout[o];
#pragma unroll 8
      for (int c2 = 0; c2 < HID; c2++)
        acc += bf2f(ht[r * HID + c2]) * sWout[o * HID + c2];
      out[(size_t)b * N_OUT + o] = acc;
    }
  }
}

extern "C" void kernel_launch(void* const* d_in, const int* in_sizes, int n_in,
                              void* d_out, int out_size, void* d_ws,
                              size_t ws_size, hipStream_t stream) {
  (void)in_sizes; (void)n_in; (void)out_size; (void)d_ws; (void)ws_size;
  const float* x    = (const float*)d_in[0];
  const float* Wih  = (const float*)d_in[1];
  const float* Whh  = (const float*)d_in[2];
  const float* bih  = (const float*)d_in[3];
  const float* bhh  = (const float*)d_in[4];
  const float* Wout = (const float*)d_in[5];
  const float* bout = (const float*)d_in[6];
  float* out = (float*)d_out;

  const int tiles  = (B_BATCH + 15) / 16;  // 6250 wave-tiles
  const int blocks = (tiles + 3) / 4;      // 4 waves (128 threads) per block

  hipLaunchKernelGGL(gru_wmma_kernel, dim3(blocks), dim3(128), 0, stream, x,
                     Wih, Whh, bih, bhh, Wout, bout, out);
}
